// LapDeepModel_64768106824129
// MI455X (gfx1250) — compile-verified, matrix operand order checked by reading, not run on hardware
//
#include <hip/hip_runtime.h>
#include <math.h>

// ---------------------------------------------------------------------------
// LapDeepModel for MI455X (gfx1250, wave32, WMMA)
//  N=50000 nodes, E=600000 edges, C=128, 15 layers x 2 inner steps.
//  GEMMs run as bf16 WMMA (v_wmma_f32_16x16x32_bf16) with f32 accumulate.
// ---------------------------------------------------------------------------

#define N_NODES 50000
#define N_EDGES 600000
#define CH      128
#define N_LAYERS 15

typedef __attribute__((ext_vector_type(16))) __bf16 v16bf;
typedef __attribute__((ext_vector_type(8)))  float  v8f;

// ---- bf16 bit helpers (round-to-nearest-even) -----------------------------
__device__ __forceinline__ unsigned short f2bf(float f) {
  union { float f; unsigned u; } x; x.f = f;
  unsigned r = x.u + 0x7FFFu + ((x.u >> 16) & 1u);
  return (unsigned short)(r >> 16);
}
__device__ __forceinline__ float bf2f(unsigned short h) {
  union { unsigned u; float f; } x; x.u = ((unsigned)h) << 16; return x.f;
}

// ---- weight prep: f32 [30,256,128] -> bf16 transposed [30,128,256] --------
__global__ void __launch_bounds__(256)
wprep_kernel(const float* __restrict__ W, unsigned short* __restrict__ Wt, int total) {
  int idx = blockIdx.x * 256 + threadIdx.x;
  if (idx >= total) return;
  int l = idx / (256 * CH);
  int rem = idx % (256 * CH);
  int k = rem / CH;
  int n = rem % CH;
  Wt[(size_t)l * 256 * CH + (size_t)n * 256 + k] = f2bf(W[idx]);
}

// ---- conv1: [N,3] @ [3,128] + b ------------------------------------------
__global__ void __launch_bounds__(256)
conv1_kernel(const float* __restrict__ in, const float* __restrict__ W,
             const float* __restrict__ b, float* __restrict__ x) {
  int idx = blockIdx.x * 256 + threadIdx.x;
  if (idx >= N_NODES * CH) return;
  int n = idx >> 7, c = idx & (CH - 1);
  float v = in[n * 3 + 0] * W[0 * CH + c]
          + in[n * 3 + 1] * W[1 * CH + c]
          + in[n * 3 + 2] * W[2 * CH + c] + b[c];
  x[idx] = v;
}

// ---- elu + pack activations to bf16 (first half of H), save res, zero prop
__global__ void __launch_bounds__(256)
elu_pack_kernel(const float* __restrict__ x, unsigned short* __restrict__ H,
                float* __restrict__ res, float* __restrict__ prop,
                int save_res, int zero_prop) {
  int idx = blockIdx.x * 256 + threadIdx.x;
  if (idx >= N_NODES * CH) return;
  int n = idx >> 7, c = idx & (CH - 1);
  float v = x[idx];
  float e = (v > 0.0f) ? v : expm1f(v);
  H[(size_t)n * 256 + c] = f2bf(e);
  if (save_res)  res[idx]  = v;
  if (zero_prop) prop[idx] = 0.0f;
}

// ---- SpMM: one wave32 per edge, 4 channels per lane, f32 atomics ----------
__global__ void __launch_bounds__(256)
spmm_kernel(const int* __restrict__ row, const int* __restrict__ col,
            const float* __restrict__ val, const unsigned short* __restrict__ H,
            float* __restrict__ prop) {
  int wid  = (blockIdx.x * 256 + threadIdx.x) >> 5;
  int lane = threadIdx.x & 31;
  if (wid >= N_EDGES) return;
  int r = row[wid], c = col[wid];
  float v = val[wid];
  const unsigned short* hc = H + (size_t)c * 256 + lane * 4;
  float* pr = prop + (size_t)r * CH + lane * 4;
  uint2 packed = *(const uint2*)hc;  // 4 bf16 values
  unsigned short h0 = (unsigned short)(packed.x & 0xFFFFu);
  unsigned short h1 = (unsigned short)(packed.x >> 16);
  unsigned short h2 = (unsigned short)(packed.y & 0xFFFFu);
  unsigned short h3 = (unsigned short)(packed.y >> 16);
  __hip_atomic_fetch_add(pr + 0, v * bf2f(h0), __ATOMIC_RELAXED, __HIP_MEMORY_SCOPE_AGENT);
  __hip_atomic_fetch_add(pr + 1, v * bf2f(h1), __ATOMIC_RELAXED, __HIP_MEMORY_SCOPE_AGENT);
  __hip_atomic_fetch_add(pr + 2, v * bf2f(h2), __ATOMIC_RELAXED, __HIP_MEMORY_SCOPE_AGENT);
  __hip_atomic_fetch_add(pr + 3, v * bf2f(h3), __ATOMIC_RELAXED, __HIP_MEMORY_SCOPE_AGENT);
}

// ---- pack SpMM result into second half of H -------------------------------
__global__ void __launch_bounds__(256)
pack_prop_kernel(const float* __restrict__ prop, unsigned short* __restrict__ H) {
  int idx = blockIdx.x * 256 + threadIdx.x;
  if (idx >= N_NODES * CH) return;
  int n = idx >> 7, c = idx & (CH - 1);
  H[(size_t)n * 256 + 128 + c] = f2bf(prop[idx]);
}

// ---- channel-wise global mean of elu(x) (reads bf16 H) --------------------
__global__ void __launch_bounds__(256)
mean_kernel(const unsigned short* __restrict__ H, float* __restrict__ mean) {
  int c = blockIdx.x;  // 0..127
  float s = 0.0f;
  for (int n = threadIdx.x; n < N_NODES; n += 256)
    s += bf2f(H[(size_t)n * 256 + c]);
  __shared__ float sm[256];
  sm[threadIdx.x] = s;
  __syncthreads();
  for (int off = 128; off > 0; off >>= 1) {
    if (threadIdx.x < off) sm[threadIdx.x] += sm[threadIdx.x + off];
    __syncthreads();
  }
  if (threadIdx.x == 0) mean[c] = sm[0] / (float)N_NODES;
}

// ---- broadcast mean into second half of H ---------------------------------
__global__ void __launch_bounds__(256)
pack_mean_kernel(const float* __restrict__ mean, unsigned short* __restrict__ H) {
  int idx = blockIdx.x * 256 + threadIdx.x;
  if (idx >= N_NODES * CH) return;
  int n = idx >> 7, c = idx & (CH - 1);
  H[(size_t)n * 256 + 128 + c] = f2bf(mean[c]);
}

// ---- WMMA GEMM: [N,256]bf16 @ [256,128]bf16 + bias (+res) -> [N,128]f32 ---
// Block = 256 threads = 8 waves; block computes 16 rows x 128 cols.
// Wave w computes the 16x16 tile at columns [16w,16w+16), K=256 in 8 WMMA steps.
// Weight tile (Wt, [128 cols][256 K] bf16) staged in 64KB LDS.
__global__ void __launch_bounds__(256)
gemm_kernel(const unsigned short* __restrict__ H,   // [N,256] bf16 bits
            const unsigned short* __restrict__ Wt,  // [128,256] bf16 bits (this layer)
            const float* __restrict__ bias,         // [128]
            const float* __restrict__ res,          // [N,128] or unused
            float* __restrict__ out,                // [N,128]
            int add_res) {
  __shared__ __align__(16) unsigned short ldsW[CH * 256];  // 65536 bytes
  int tid = threadIdx.x;

  // cooperative 128-bit copy of the weight tile: 4096 uint4 / 256 threads
  const uint4* gw = (const uint4*)Wt;
  uint4* lw = (uint4*)ldsW;
#pragma unroll
  for (int r = 0; r < 16; ++r) lw[tid + 256 * r] = gw[tid + 256 * r];
  __syncthreads();

  int wave = tid >> 5;
  int lane = tid & 31;
  int half = lane >> 4;            // K-subgroup select
  int m0 = blockIdx.x * 16;
  int mrow = m0 + (lane & 15);     // A: lane carries one M row
  int ncol = wave * 16 + (lane & 15);  // B: lane carries one N column

  const unsigned short* arow = H + (size_t)mrow * 256 + 8 * half;
  const unsigned short* brow = ldsW + (size_t)ncol * 256 + 8 * half;

  v8f acc = {};
#pragma unroll
  for (int kk = 0; kk < 256; kk += 32) {
    union { uint4 q[2]; v16bf v; } A, B;
    // elements 0..7 = K kk..kk+7(+8*half), elements 8..15 = K kk+16..kk+23(+8*half)
    A.q[0] = *(const uint4*)(arow + kk);
    A.q[1] = *(const uint4*)(arow + kk + 16);
    B.q[0] = *(const uint4*)(brow + kk);
    B.q[1] = *(const uint4*)(brow + kk + 16);
    acc = __builtin_amdgcn_wmma_f32_16x16x32_bf16(
        /*neg_a=*/false, A.v, /*neg_b=*/false, B.v,
        /*c_mod=*/(short)0, acc, /*reuse_a=*/false, /*reuse_b=*/false);
  }

  float bv = bias[ncol];
#pragma unroll
  for (int r = 0; r < 8; ++r) {
    int m = m0 + r + 8 * half;     // C/D layout: VGPR r -> M=r (lanes 0-15), M=r+8 (16-31)
    float v = acc[r] + bv;
    size_t o = (size_t)m * CH + ncol;
    if (add_res) v += res[o];
    out[o] = v;
  }
}

// ---- final: elu(x) @ conv2_W + conv2_b + inputs[:, :, 0] ------------------
__global__ void __launch_bounds__(256)
final_kernel(const float* __restrict__ x, const float* __restrict__ w2,
             const float* __restrict__ b2, const float* __restrict__ in,
             float* __restrict__ out) {
  int wid  = (blockIdx.x * 256 + threadIdx.x) >> 5;
  int lane = threadIdx.x & 31;
  if (wid >= N_NODES) return;
  float s = 0.0f;
  for (int c = lane; c < CH; c += 32) {
    float v = x[(size_t)wid * CH + c];
    float e = (v > 0.0f) ? v : expm1f(v);
    s += e * w2[c];
  }
#pragma unroll
  for (int off = 16; off > 0; off >>= 1) s += __shfl_xor(s, off, 32);
  if (lane == 0) out[wid] = s + b2[0] + in[wid * 3 + 0];
}

// ---------------------------------------------------------------------------
extern "C" void kernel_launch(void* const* d_in, const int* in_sizes, int n_in,
                              void* d_out, int out_size, void* d_ws, size_t ws_size,
                              hipStream_t stream) {
  (void)in_sizes; (void)n_in; (void)out_size; (void)ws_size;
  const int*   L_row    = (const int*)d_in[0];
  const int*   L_col    = (const int*)d_in[1];
  const float* L_val    = (const float*)d_in[2];
  // d_in[3] = mask (all ones; mean uses N directly)
  const float* inputs   = (const float*)d_in[4];
  const float* conv1_W  = (const float*)d_in[5];
  const float* conv1_b  = (const float*)d_in[6];
  const float* blocks_W = (const float*)d_in[7];
  const float* blocks_b = (const float*)d_in[8];
  const float* conv2_W  = (const float*)d_in[9];
  const float* conv2_b  = (const float*)d_in[10];

  // workspace carve-up (256B aligned)
  char* ws = (char*)d_ws;
  size_t o = 0;
  auto alignup = [](size_t v) { return (v + 255) & ~(size_t)255; };
  float* xf            = (float*)(ws + o);          o = alignup(o + (size_t)N_NODES * CH * 4);
  float* resb          = (float*)(ws + o);          o = alignup(o + (size_t)N_NODES * CH * 4);
  float* propf         = (float*)(ws + o);          o = alignup(o + (size_t)N_NODES * CH * 4);
  unsigned short* Hbf  = (unsigned short*)(ws + o); o = alignup(o + (size_t)N_NODES * 256 * 2);
  unsigned short* Wtbf = (unsigned short*)(ws + o); o = alignup(o + (size_t)N_LAYERS * 2 * 256 * CH * 2);
  float* meanb         = (float*)(ws + o);          o = alignup(o + 128 * 4);

  const int nc_blocks   = (N_NODES * CH + 255) / 256;     // 25000
  const int gemm_blocks = N_NODES / 16;                   // 3125 (exact)
  const int spmm_blocks = (N_EDGES + 7) / 8;              // 75000 (8 waves/block)
  const int wtotal      = N_LAYERS * 2 * 256 * CH;

  // one-time prep (re-run every call: deterministic, no cached state)
  wprep_kernel<<<(wtotal + 255) / 256, 256, 0, stream>>>(blocks_W, Wtbf, wtotal);
  conv1_kernel<<<nc_blocks, 256, 0, stream>>>(inputs, conv1_W, conv1_b, xf);

  for (int i = 0; i < N_LAYERS; ++i) {
    for (int j = 0; j < 2; ++j) {
      int even = (i % 2 == 0);
      elu_pack_kernel<<<nc_blocks, 256, 0, stream>>>(xf, Hbf, resb, propf,
                                                     /*save_res=*/(j == 0),
                                                     /*zero_prop=*/even);
      if (even) {
        spmm_kernel<<<spmm_blocks, 256, 0, stream>>>(L_row, L_col, L_val, Hbf, propf);
        pack_prop_kernel<<<nc_blocks, 256, 0, stream>>>(propf, Hbf);
      } else {
        mean_kernel<<<CH, 256, 0, stream>>>(Hbf, meanb);
        pack_mean_kernel<<<nc_blocks, 256, 0, stream>>>(meanb, Hbf);
      }
      const unsigned short* Wl = Wtbf + (size_t)(i * 2 + j) * 256 * CH;
      const float* bl = blocks_b + (size_t)(i * 2 + j) * CH;
      gemm_kernel<<<gemm_blocks, 256, 0, stream>>>(Hbf, Wl, bl, resb, xf,
                                                   /*add_res=*/(j == 1));
    }
  }

  final_kernel<<<(N_NODES * 32 + 255) / 256, 256, 0, stream>>>(
      xf, conv2_W, conv2_b, inputs, (float*)d_out);
}
